// NOVAE_49967649521705
// MI455X (gfx1250) — matrix operand order repaired and checked by reading, not compiled, not run on hardware
//
#include <hip/hip_runtime.h>
#include <hip/hip_bf16.h>

// ---------------------------------------------------------------------------
// Problem constants (shapes fixed by reference setup_inputs)
// ---------------------------------------------------------------------------
#define BATCH   4096
#define NPRIOR  4096
#define LAT     128
#define DIN     784
#define SINK_ITERS 1000
#define SINK_REG   0.05f
#define SBLK    256                    // persistent Sinkhorn blocks
#define ROWS_PER_BLK (4096 / SBLK)     // 16 rows/block, 2 rows/wave

typedef float  v8f    __attribute__((ext_vector_type(8)));
typedef __bf16 v16bf  __attribute__((ext_vector_type(16)));
typedef __bf16 bf16x8 __attribute__((ext_vector_type(8)));

// ---------------------------------------------------------------------------
// Elementwise convert f32 -> bf16
// ---------------------------------------------------------------------------
__global__ void k_f32_to_bf16(const float* __restrict__ in, __bf16* __restrict__ out, int n) {
    int i = blockIdx.x * 256 + threadIdx.x;
    if (i < n) out[i] = (__bf16)in[i];
}

// z_prior (4096x128 f32) -> transposed bf16 (128x4096)
__global__ void k_zp_transpose(const float* __restrict__ zp, __bf16* __restrict__ zpT) {
    int idx = blockIdx.x * 256 + threadIdx.x;       // 128*4096 total
    int k = idx >> 12;                              // 0..127
    int n = idx & 4095;                             // 0..4095
    zpT[idx] = (__bf16)zp[n * LAT + k];
}

// init u, v = 1/B ; gmax = 0 ; barrier state = 0
__global__ void k_sink_init(float* __restrict__ u, float* __restrict__ v,
                            unsigned* __restrict__ gmax,
                            unsigned* __restrict__ counter, unsigned* __restrict__ gen) {
    int i = blockIdx.x * 256 + threadIdx.x;
    if (i < BATCH) { u[i] = 1.0f / BATCH; v[i] = 1.0f / NPRIOR; }
    if (i == 0) { *gmax = 0u; *counter = 0u; *gen = 0u; }
}

// row squared norms of (rows x 128) f32 matrix: one wave per row
__global__ __launch_bounds__(256) void k_row_norm_sq(const float* __restrict__ X, float* __restrict__ out) {
    int lane = threadIdx.x & 31, wave = threadIdx.x >> 5;
    int row = blockIdx.x * 8 + wave;
    const float* xr = X + (size_t)row * LAT;
    float s = 0.0f;
    #pragma unroll
    for (int c = 0; c < 4; ++c) { float t = xr[lane + 32 * c]; s += t * t; }
    #pragma unroll
    for (int off = 16; off; off >>= 1) s += __shfl_xor(s, off, 32);
    if (lane == 0) out[row] = s;
}

// ---------------------------------------------------------------------------
// bf16 WMMA GEMM:  C(MxN) = act(A(MxK) @ B(KxN) + bias)
// block tile 128x128, 256 threads = 8 waves (2 M-waves x 4 N-waves),
// per-wave 64x32 = 4x2 tiles of v_wmma_f32_16x16x32_bf16.
// ALIGNED path: unguarded b128 staging, double-buffered LDS, 1 barrier/Kstep.
// ---------------------------------------------------------------------------
#define GBM 128
#define GBN 128
#define GBK 32
#define GKP 40   // padded K pitch in LDS (elements); 80B rows -> 16B-aligned chunks

template <bool RELU, bool ALIGNED>
__global__ __launch_bounds__(256)
void k_gemm_bf16(const __bf16* __restrict__ A, const __bf16* __restrict__ B,
                 const float* __restrict__ bias,
                 float* __restrict__ Cf, __bf16* __restrict__ Cb,
                 int M, int N, int K) {
    __shared__ __bf16 As[2 * GBM * GKP];    // [buf][row m][k]
    __shared__ __bf16 BsT[2 * GBN * GKP];   // [buf][col n][k]

    const int tid   = threadIdx.x;
    const int lane  = tid & 31;
    const int wave  = tid >> 5;
    const int waveM = wave >> 2;        // 0..1
    const int waveN = wave & 3;         // 0..3
    const int m0 = blockIdx.y * GBM;
    const int n0 = blockIdx.x * GBN;

    v8f acc[4][2];
    #pragma unroll
    for (int mt = 0; mt < 4; ++mt)
        #pragma unroll
        for (int nt = 0; nt < 2; ++nt)
            #pragma unroll
            for (int i = 0; i < 8; ++i) acc[mt][nt][i] = 0.0f;

    const int l  = lane & 15;
    const int h  = lane >> 4;
    const int kb = h * 8;       // A fragment K base (ISA 16-bit A layout)
    const int ks = h * 8;       // B fragment K base (same table order as A)

    if (ALIGNED) {
        const int ar  = tid >> 1;            // A: row in tile      (0..127)
        const int akc = (tid & 1) * 16;      // A: k chunk          (0 / 16)
        const int bkr = (tid >> 4) << 1;     // B: k row pair base  (0..30)
        const int bnc = (tid & 15) * 8;      // B: n chunk          (0..120)
        const __bf16* aSrc = A + (size_t)(m0 + ar) * K + akc;
        const __bf16* bSrc = B + (size_t)bkr * N + n0 + bnc;

        bf16x8 ra0, ra1, rb0, rb1;
        auto loadTile = [&](int k0) {
            ra0 = *(const bf16x8*)(aSrc + k0);
            ra1 = *(const bf16x8*)(aSrc + k0 + 8);
            const __bf16* bs = bSrc + (size_t)k0 * N;
            rb0 = *(const bf16x8*)bs;            // row k+bkr
            rb1 = *(const bf16x8*)(bs + N);      // row k+bkr+1
        };
        auto storeTile = [&](int buf) {
            __bf16* as = As + buf * (GBM * GKP);
            *(bf16x8*)&as[ar * GKP + akc]     = ra0;
            *(bf16x8*)&as[ar * GKP + akc + 8] = ra1;
            __bf16* bs = BsT + buf * (GBN * GKP);
            #pragma unroll
            for (int i = 0; i < 8; ++i) {        // packed b32 transpose scatter
                union { __bf16 hh[2]; unsigned u; } pk;
                pk.hh[0] = rb0[i]; pk.hh[1] = rb1[i];
                *(unsigned*)&bs[(bnc + i) * GKP + bkr] = pk.u;
            }
        };

        loadTile(0);
        storeTile(0);
        int buf = 0;
        for (int k0 = 0; k0 < K; k0 += GBK) {
            const bool notLast = (k0 + GBK < K);
            if (notLast) loadTile(k0 + GBK);   // overlap VMEM with barrier+math
            __syncthreads();

            const __bf16* as = As + buf * (GBM * GKP);
            const __bf16* bs = BsT + buf * (GBN * GKP);
            v16bf af[4], bfr[2];
            #pragma unroll
            for (int mt = 0; mt < 4; ++mt) {
                int m = waveM * 64 + mt * 16 + l;
                bf16x8 lo = *(const bf16x8*)&as[m * GKP + kb];
                bf16x8 hi = *(const bf16x8*)&as[m * GKP + kb + 16];
                #pragma unroll
                for (int i = 0; i < 8; ++i) { af[mt][i] = lo[i]; af[mt][8 + i] = hi[i]; }
            }
            #pragma unroll
            for (int nt = 0; nt < 2; ++nt) {
                int n = waveN * 32 + nt * 16 + l;
                bf16x8 lo = *(const bf16x8*)&bs[n * GKP + ks];
                bf16x8 hi = *(const bf16x8*)&bs[n * GKP + ks + 16];
                #pragma unroll
                for (int i = 0; i < 8; ++i) { bfr[nt][i] = lo[i]; bfr[nt][8 + i] = hi[i]; }
            }
            #pragma unroll
            for (int mt = 0; mt < 4; ++mt)
                #pragma unroll
                for (int nt = 0; nt < 2; ++nt)
                    acc[mt][nt] = __builtin_amdgcn_wmma_f32_16x16x32_bf16(
                        false, af[mt], false, bfr[nt], (short)0, acc[mt][nt], false, false);

            if (notLast) storeTile(buf ^ 1);   // safe: peers read buf, write buf^1
            buf ^= 1;
        }
    } else {
        // ---- ragged path: chunk-level guards, vector fast chunks ---------
        for (int k0 = 0; k0 < K; k0 += GBK) {
            {   // A tile
                int r  = tid >> 1;
                int kc = (tid & 1) * 16;
                int gm = m0 + r;
                const __bf16* src = A + (size_t)gm * K + k0 + kc;
                __bf16* dst = &As[r * GKP + kc];
                if (gm < M && k0 + kc + 16 <= K) {
                    *(bf16x8*)dst       = *(const bf16x8*)src;
                    *(bf16x8*)(dst + 8) = *(const bf16x8*)(src + 8);
                } else {
                    for (int i = 0; i < 16; ++i) {
                        int k = k0 + kc + i;
                        dst[i] = (gm < M && k < K) ? src[i] : (__bf16)0.0f;
                    }
                }
            }
            {   // B tile (transposed into BsT)
                int kr = tid >> 3;
                int nc = (tid & 7) * 16;
                int gk = k0 + kr;
                const __bf16* src = B + (size_t)gk * N + n0 + nc;
                if (gk < K && n0 + nc + 16 <= N) {
                    bf16x8 lo = *(const bf16x8*)src;
                    bf16x8 hi = *(const bf16x8*)(src + 8);
                    #pragma unroll
                    for (int i = 0; i < 8; ++i) {
                        BsT[(nc + i) * GKP + kr]     = lo[i];
                        BsT[(nc + 8 + i) * GKP + kr] = hi[i];
                    }
                } else {
                    for (int i = 0; i < 16; ++i) {
                        int gn = n0 + nc + i;
                        BsT[(nc + i) * GKP + kr] = (gk < K && gn < N) ? src[i] : (__bf16)0.0f;
                    }
                }
            }
            __syncthreads();

            v16bf af[4], bfr[2];
            #pragma unroll
            for (int mt = 0; mt < 4; ++mt) {
                int m = waveM * 64 + mt * 16 + l;
                bf16x8 lo = *(const bf16x8*)&As[m * GKP + kb];
                bf16x8 hi = *(const bf16x8*)&As[m * GKP + kb + 16];
                #pragma unroll
                for (int i = 0; i < 8; ++i) { af[mt][i] = lo[i]; af[mt][8 + i] = hi[i]; }
            }
            #pragma unroll
            for (int nt = 0; nt < 2; ++nt) {
                int n = waveN * 32 + nt * 16 + l;
                bf16x8 lo = *(const bf16x8*)&BsT[n * GKP + ks];
                bf16x8 hi = *(const bf16x8*)&BsT[n * GKP + ks + 16];
                #pragma unroll
                for (int i = 0; i < 8; ++i) { bfr[nt][i] = lo[i]; bfr[nt][8 + i] = hi[i]; }
            }
            #pragma unroll
            for (int mt = 0; mt < 4; ++mt)
                #pragma unroll
                for (int nt = 0; nt < 2; ++nt)
                    acc[mt][nt] = __builtin_amdgcn_wmma_f32_16x16x32_bf16(
                        false, af[mt], false, bfr[nt], (short)0, acc[mt][nt], false, false);
            __syncthreads();
        }
    }

    // --- epilogue: bias + relu + store (f32 and/or bf16) -----------------
    #pragma unroll
    for (int nt = 0; nt < 2; ++nt) {
        int gn = n0 + waveN * 32 + nt * 16 + l;
        float bv = (bias != nullptr && (ALIGNED || gn < N)) ? bias[gn] : 0.0f;
        #pragma unroll
        for (int mt = 0; mt < 4; ++mt) {
            int mBase = m0 + waveM * 64 + mt * 16 + h * 8;
            #pragma unroll
            for (int r = 0; r < 8; ++r) {
                int gm = mBase + r;
                if (ALIGNED || (gm < M && gn < N)) {
                    float vv = acc[mt][nt][r] + bv;
                    if (RELU) vv = fmaxf(vv, 0.0f);
                    if (Cf) Cf[(size_t)gm * N + gn] = vv;
                    if (Cb) Cb[(size_t)gm * N + gn] = (__bf16)vv;
                }
            }
        }
    }
}

// ---------------------------------------------------------------------------
// cost[i,j] = max(rn[i] + cn[j] - 2*G[i,j], 0), plus global max via atomicMax
// ---------------------------------------------------------------------------
__global__ __launch_bounds__(256)
void k_cost_and_max(const float* __restrict__ G, const float* __restrict__ rn,
                    const float* __restrict__ cn, float* __restrict__ cost,
                    unsigned* __restrict__ gmax) {
    __shared__ float red[256];
    size_t base = (size_t)blockIdx.x * 4096;
    float mx = 0.0f;
    #pragma unroll 4
    for (int k = 0; k < 16; ++k) {
        size_t idx = base + (size_t)k * 256 + threadIdx.x;
        int i = (int)(idx >> 12);
        int j = (int)(idx & 4095);
        float c = fmaxf(rn[i] + cn[j] - 2.0f * G[idx], 0.0f);
        cost[idx] = c;
        mx = fmaxf(mx, c);
    }
    red[threadIdx.x] = mx;
    __syncthreads();
    for (int s = 128; s; s >>= 1) {
        if (threadIdx.x < s) red[threadIdx.x] = fmaxf(red[threadIdx.x], red[threadIdx.x + s]);
        __syncthreads();
    }
    if (threadIdx.x == 0) atomicMax(gmax, __float_as_uint(red[0]));
}

// ---------------------------------------------------------------------------
// K = exp(-cost / (reg*(max+1e-12))) as bf16, plus transposed copy (LDS-tiled)
// ---------------------------------------------------------------------------
__global__ __launch_bounds__(256)
void k_build_K(const float* __restrict__ cost, const unsigned* __restrict__ gmax,
               __bf16* __restrict__ Kb, __bf16* __restrict__ KbT) {
    __shared__ __bf16 tile[64 * 65];
    float mx = __uint_as_float(*gmax);
    float sc = -1.0f / (SINK_REG * (mx + 1e-12f));
    int i0 = (blockIdx.x >> 6) * 64;
    int j0 = (blockIdx.x & 63) * 64;
    #pragma unroll 4
    for (int k = 0; k < 16; ++k) {
        int lin = k * 256 + threadIdx.x;
        int r = lin >> 6, c = lin & 63;
        size_t idx = (size_t)(i0 + r) * 4096 + j0 + c;
        __bf16 e = (__bf16)expf(cost[idx] * sc);
        Kb[idx] = e;
        tile[r * 65 + c] = e;
    }
    __syncthreads();
    #pragma unroll 4
    for (int k = 0; k < 16; ++k) {
        int lin = k * 256 + threadIdx.x;
        int r = lin >> 6, c = lin & 63;
        KbT[(size_t)(j0 + r) * 4096 + i0 + c] = tile[c * 65 + r];
    }
}

// ---------------------------------------------------------------------------
// Software grid barrier (all SBLK blocks co-resident; light kernel)
// ---------------------------------------------------------------------------
__device__ __forceinline__ void grid_sync(unsigned* counter, unsigned* gen, int tid) {
    __syncthreads();
    if (tid == 0) {
        unsigned g = __hip_atomic_load(gen, __ATOMIC_RELAXED, __HIP_MEMORY_SCOPE_AGENT);
        __threadfence();  // publish this block's writes (after __syncthreads)
        unsigned prev = __hip_atomic_fetch_add(counter, 1u, __ATOMIC_ACQ_REL, __HIP_MEMORY_SCOPE_AGENT);
        if (prev == (unsigned)(SBLK - 1)) {
            __hip_atomic_store(counter, 0u, __ATOMIC_RELAXED, __HIP_MEMORY_SCOPE_AGENT);
            __hip_atomic_fetch_add(gen, 1u, __ATOMIC_RELEASE, __HIP_MEMORY_SCOPE_AGENT);
        } else {
            while (__hip_atomic_load(gen, __ATOMIC_ACQUIRE, __HIP_MEMORY_SCOPE_AGENT) == g)
                __builtin_amdgcn_s_sleep(4);
        }
    }
    __syncthreads();
}

// One matvec phase: xout[row] = marg / dot(Km[row,:], xin)  for this block's rows
__device__ __forceinline__ void sink_phase(const __bf16* __restrict__ Km,
                                           const float* __restrict__ xin,
                                           float* __restrict__ xout,
                                           float marg, float* vs, int tid) {
    #pragma unroll
    for (int j = tid; j < 4096; j += 256) vs[j] = xin[j];
    __syncthreads();
    int lane = tid & 31, wave = tid >> 5;
    int row0 = blockIdx.x * ROWS_PER_BLK + wave * 2;    // 2 rows per wave
    const __bf16* kr0 = Km + (size_t)row0 * 4096;
    const __bf16* kr1 = kr0 + 4096;
    float a0 = 0.0f, a1 = 0.0f;
    #pragma unroll 4
    for (int c = 0; c < 16; ++c) {
        int j = c * 256 + lane * 8;
        bf16x8 k0 = *(const bf16x8*)(kr0 + j);
        bf16x8 k1 = *(const bf16x8*)(kr1 + j);
        const float4* vp = (const float4*)&vs[j];
        float4 v0 = vp[0], v1 = vp[1];   // shared by both rows
        a0 += (float)k0[0] * v0.x + (float)k0[1] * v0.y + (float)k0[2] * v0.z + (float)k0[3] * v0.w
            + (float)k0[4] * v1.x + (float)k0[5] * v1.y + (float)k0[6] * v1.z + (float)k0[7] * v1.w;
        a1 += (float)k1[0] * v0.x + (float)k1[1] * v0.y + (float)k1[2] * v0.z + (float)k1[3] * v0.w
            + (float)k1[4] * v1.x + (float)k1[5] * v1.y + (float)k1[6] * v1.z + (float)k1[7] * v1.w;
    }
    #pragma unroll
    for (int off = 16; off; off >>= 1) {
        a0 += __shfl_xor(a0, off, 32);
        a1 += __shfl_xor(a1, off, 32);
    }
    if (lane == 0) { xout[row0] = marg / a0; xout[row0 + 1] = marg / a1; }
}

// Persistent Sinkhorn: all 1000 iterations in one launch, grid-synced phases.
__global__ __launch_bounds__(256)
void k_sinkhorn_persistent(const __bf16* __restrict__ Kb, const __bf16* __restrict__ KbT,
                           float* __restrict__ u, float* __restrict__ v, float marg,
                           unsigned* counter, unsigned* gen) {
    __shared__ __align__(16) float vs[4096];
    const int tid = threadIdx.x;
    for (int it = 0; it < SINK_ITERS; ++it) {
        sink_phase(KbT, u, v, marg, vs, tid);   // v = b / (K^T u)
        grid_sync(counter, gen, tid);
        sink_phase(Kb, v, u, marg, vs, tid);    // u = a / (K v)
        grid_sync(counter, gen, tid);
    }
}

// ---------------------------------------------------------------------------
// P[i,j] = bf16( u[i] * K[i,j] * v[j] )   (the transport plan, as GEMM A)
// ---------------------------------------------------------------------------
__global__ void k_build_P(const __bf16* __restrict__ Kb, const float* __restrict__ u,
                          const float* __restrict__ v, __bf16* __restrict__ P) {
    size_t idx = (size_t)blockIdx.x * 256 + threadIdx.x;
    int i = (int)(idx >> 12), j = (int)(idx & 4095);
    P[idx] = (__bf16)(u[i] * (float)Kb[idx] * v[j]);
}

// ---------------------------------------------------------------------------
// Host-side launcher
// ---------------------------------------------------------------------------
static inline size_t alignup(size_t x) { return (x + 255) & ~(size_t)255; }

static void launch_gemm(const __bf16* A, const __bf16* B, const float* bias,
                        float* Cf, __bf16* Cb, int M, int N, int K, bool relu,
                        hipStream_t s) {
    dim3 grid((N + GBN - 1) / GBN, (M + GBM - 1) / GBM);
    bool aligned = (M % GBM == 0) && (N % GBN == 0) && (K % GBK == 0);
    if (aligned) {
        if (relu) k_gemm_bf16<true , true ><<<grid, 256, 0, s>>>(A, B, bias, Cf, Cb, M, N, K);
        else      k_gemm_bf16<false, true ><<<grid, 256, 0, s>>>(A, B, bias, Cf, Cb, M, N, K);
    } else {
        if (relu) k_gemm_bf16<true , false><<<grid, 256, 0, s>>>(A, B, bias, Cf, Cb, M, N, K);
        else      k_gemm_bf16<false, false><<<grid, 256, 0, s>>>(A, B, bias, Cf, Cb, M, N, K);
    }
}

extern "C" void kernel_launch(void* const* d_in, const int* in_sizes, int n_in,
                              void* d_out, int out_size, void* d_ws, size_t ws_size,
                              hipStream_t stream) {
    (void)in_sizes; (void)n_in; (void)out_size; (void)ws_size;

    const float* x_f32  = (const float*)d_in[0];
    const float* zp_f32 = (const float*)d_in[1];
    const float* encW[4] = {(const float*)d_in[2], (const float*)d_in[4], (const float*)d_in[6], (const float*)d_in[8]};
    const float* encB[4] = {(const float*)d_in[3], (const float*)d_in[5], (const float*)d_in[7], (const float*)d_in[9]};
    const float* decW[4] = {(const float*)d_in[10], (const float*)d_in[12], (const float*)d_in[14], (const float*)d_in[16]};
    const float* decB[4] = {(const float*)d_in[11], (const float*)d_in[13], (const float*)d_in[15], (const float*)d_in[17]};

    const int encD[5] = {DIN, 1024, 2048, 1024, LAT};
    const int decD[5] = {LAT, 1024, 2048, 1024, DIN};

    // ---- workspace carve-up ------------------------------------------------
    char* p = (char*)d_ws;
    auto take = [&](size_t bytes) { char* r = p; p += alignup(bytes); return r; };

    __bf16* xbf   = (__bf16*)take((size_t)BATCH * DIN * 2);         // 6.4 MB
    __bf16* actA  = (__bf16*)take((size_t)BATCH * 2048 * 2);        // 16.8 MB
    __bf16* actB  = (__bf16*)take((size_t)BATCH * 2048 * 2);        // 16.8 MB
    __bf16* Wbf[8];
    size_t wElems[8];
    for (int i = 0; i < 4; ++i) wElems[i]     = (size_t)encD[i] * encD[i + 1];
    for (int i = 0; i < 4; ++i) wElems[4 + i] = (size_t)decD[i] * decD[i + 1];
    for (int i = 0; i < 8; ++i) Wbf[i] = (__bf16*)take(wElems[i] * 2);  // 20.5 MB
    float*  Gcost = (float*)take((size_t)BATCH * NPRIOR * 4);       // 67 MB (Gram->cost->P)
    __bf16* Kb    = (__bf16*)take((size_t)BATCH * NPRIOR * 2);      // 33.5 MB
    __bf16* KbT   = (__bf16*)take((size_t)BATCH * NPRIOR * 2);      // 33.5 MB
    float*  zf32  = (float*)take((size_t)BATCH * LAT * 4);          // 2 MB
    __bf16* zpT   = (__bf16*)take((size_t)LAT * NPRIOR * 2);        // 1 MB
    __bf16* zpbf  = (__bf16*)take((size_t)NPRIOR * LAT * 2);        // 1 MB
    float*  rn    = (float*)take(BATCH * 4);
    float*  cn    = (float*)take(NPRIOR * 4);
    float*  uvec  = (float*)take(BATCH * 4);
    float*  vvec  = (float*)take(NPRIOR * 4);
    unsigned* gmax    = (unsigned*)take(256);
    unsigned* counter = (unsigned*)take(256);
    unsigned* gen     = (unsigned*)take(256);
    __bf16* Pbf   = (__bf16*)Gcost;   // reuse cost buffer after K is built

    // ---- 0. init + conversions --------------------------------------------
    k_sink_init<<<16, 256, 0, stream>>>(uvec, vvec, gmax, counter, gen);

    { int n = BATCH * DIN;   k_f32_to_bf16<<<(n + 255) / 256, 256, 0, stream>>>(x_f32, xbf, n); }
    for (int i = 0; i < 4; ++i) {
        int n = (int)wElems[i];
        k_f32_to_bf16<<<(n + 255) / 256, 256, 0, stream>>>(encW[i], Wbf[i], n);
    }
    for (int i = 0; i < 4; ++i) {
        int n = (int)wElems[4 + i];
        k_f32_to_bf16<<<(n + 255) / 256, 256, 0, stream>>>(decW[i], Wbf[4 + i], n);
    }
    { int n = NPRIOR * LAT;  k_f32_to_bf16<<<(n + 255) / 256, 256, 0, stream>>>(zp_f32, zpbf, n); }
    k_zp_transpose<<<(LAT * NPRIOR) / 256, 256, 0, stream>>>(zp_f32, zpT);

    // ---- 1. encoder MLP ----------------------------------------------------
    launch_gemm(xbf,  Wbf[0], encB[0], nullptr, actA, BATCH, encD[1], encD[0], true,  stream);
    launch_gemm(actA, Wbf[1], encB[1], nullptr, actB, BATCH, encD[2], encD[1], true,  stream);
    launch_gemm(actB, Wbf[2], encB[2], nullptr, actA, BATCH, encD[3], encD[2], true,  stream);
    launch_gemm(actA, Wbf[3], encB[3], zf32,    actB, BATCH, encD[4], encD[3], false, stream);  // z -> actB(bf16)+zf32

    // ---- 2. squared cdist + normalization ---------------------------------
    k_row_norm_sq<<<BATCH / 8, 256, 0, stream>>>(zf32,   rn);
    k_row_norm_sq<<<NPRIOR / 8, 256, 0, stream>>>(zp_f32, cn);
    launch_gemm(actB, zpT, nullptr, Gcost, nullptr, BATCH, NPRIOR, LAT, false, stream);  // G = z @ zp^T
    k_cost_and_max<<<4096, 256, 0, stream>>>(Gcost, rn, cn, Gcost, gmax);
    k_build_K<<<4096, 256, 0, stream>>>(Gcost, gmax, Kb, KbT);

    // ---- 3. Sinkhorn: ONE persistent kernel, internal grid barriers --------
    k_sinkhorn_persistent<<<SBLK, 256, 0, stream>>>(Kb, KbT, uvec, vvec,
                                                    1.0f / (float)BATCH, counter, gen);

    // ---- 4. z_selected = (u * K * v) @ z_prior -----------------------------
    k_build_P<<<(BATCH * (NPRIOR / 256)), 256, 0, stream>>>(Kb, uvec, vvec, Pbf);
    launch_gemm(Pbf, zpbf, nullptr, nullptr, actA, BATCH, LAT, NPRIOR, false, stream);

    // ---- 5. decoder MLP -> d_out (f32) -------------------------------------
    launch_gemm(actA, Wbf[4], decB[0], nullptr, actB, BATCH, decD[1], decD[0], true,  stream);
    launch_gemm(actB, Wbf[5], decB[1], nullptr, actA, BATCH, decD[2], decD[1], true,  stream);
    launch_gemm(actA, Wbf[6], decB[2], nullptr, actB, BATCH, decD[3], decD[2], true,  stream);
    launch_gemm(actB, Wbf[7], decB[3], (float*)d_out, nullptr, BATCH, decD[4], decD[3], false, stream);
}
